// GRU_39118562132044
// MI455X (gfx1250) — matrix operand order repaired and checked by reading, not compiled, not run on hardware
//
#include <hip/hip_runtime.h>
#include <hip/hip_bf16.h>

// ---------------------------------------------------------------------------
// GRU for MI455X (gfx1250, wave32).
//   hT over T=512 steps, B=128, D=512, H=1024.
//   Per step: one GEMM  A=[x_t | h] (K=1536)  x  B=[Wi;Wh] (1536x3072), bf16
//   WMMA with f32 accumulate; n-gate kept in split accumulators (K<512 vs >=512)
//   because the reference computes tanh(inn + r*hn).
//   A-panel (32 x 1536 bf16) staged into LDS with GLOBAL_LOAD_ASYNC_TO_LDS_B128
//   (ASYNCcnt path); each wave computes 2 M-tiles x {r,z,n-split} = 8 accums so
//   every B fragment feeds 2 WMMAs.
// ---------------------------------------------------------------------------

typedef __attribute__((ext_vector_type(16))) __bf16 v16bf;
typedef __attribute__((ext_vector_type(8)))  float  v8f;

union Frag16 {
    v16bf v;
    uint4 q[2];
};

#define T_STEPS   512
#define BATCH     128
#define DIM_IN    512
#define HID       1024
#define KTOT      1536          // DIM_IN + HID
#define NGATE     3072          // 3*HID
#define LDS_STRIDE 1544         // 1536 + 8 bf16 pad -> row stride 772 dwords (≡4 banks)

// ---------------------------------------------------------------------------
// Prologue: build combined transposed weight  Wt[n][k] (bf16, [3072][1536])
//   k <  512 : Wi[k][n] ;  k >= 512 : Wh[k-512][n]
// Row-major in n so the per-lane B-fragment read is one contiguous 32B.
// ---------------------------------------------------------------------------
__global__ __launch_bounds__(256) void build_wt(const float* __restrict__ Wi,
                                                const float* __restrict__ Wh,
                                                __bf16* __restrict__ Wt) {
    size_t i = (size_t)blockIdx.x * 256 + threadIdx.x;   // 0 .. 3072*1536-1
    int k = (int)(i % KTOT);
    int n = (int)(i / KTOT);
    float v = (k < DIM_IN) ? Wi[(size_t)k * NGATE + n]
                           : Wh[(size_t)(k - DIM_IN) * NGATE + n];
    Wt[i] = (__bf16)v;
}

// x [B][T][D] f32  ->  xbf [T][B][D] bf16  (per-step slice contiguous)
__global__ __launch_bounds__(256) void convert_x(const float* __restrict__ x,
                                                 __bf16* __restrict__ xbf) {
    size_t i = (size_t)blockIdx.x * 256 + threadIdx.x;   // 0 .. 512*128*512-1
    int d  = (int)(i % DIM_IN);
    int tb = (int)(i / DIM_IN);
    int b  = tb % BATCH;
    int t  = tb / BATCH;
    xbf[i] = (__bf16)x[((size_t)b * T_STEPS + t) * DIM_IN + d];
}

__global__ __launch_bounds__(256) void init_h(float* __restrict__ hf,
                                              __bf16* __restrict__ hb) {
    int i = blockIdx.x * 256 + threadIdx.x;              // 0 .. 128*1024-1
    hf[i] = 0.0f;
    hb[i] = (__bf16)0.0f;
}

__device__ inline float sigm(float v) { return 1.0f / (1.0f + __expf(-v)); }

#define WMMA_BF16(A, B, C) \
    __builtin_amdgcn_wmma_f32_16x16x32_bf16(false, (A), false, (B), (short)0, (C), false, false)

// ---------------------------------------------------------------------------
// One GRU timestep.
// grid = 64 blocks: rowPair = blk&3 (M=128/32), jGroup = blk>>2 (H=1024/64)
// block = 128 threads = 4 waves; wave w owns j-slice [j0, j0+16) and computes
// 2 M-tiles x 4 accumulators (r, z, in = x-part of n, hn = h-part of n).
// ---------------------------------------------------------------------------
__global__ __launch_bounds__(128) void gru_step(
    const __bf16* __restrict__ xbf,      // [T][B][512]
    const __bf16* __restrict__ Wt,       // [3072][1536]
    const float*  __restrict__ bi,       // [3072]
    const float*  __restrict__ bh,       // [3072]
    const float*  __restrict__ hprevF,   // [128][1024]
    const __bf16* __restrict__ hprevB,   // [128][1024]
    float*        __restrict__ hnextF,   // [128][1024]
    __bf16*       __restrict__ hnextB,   // [128][1024]
    int t) {
    __shared__ __bf16 ldsA[32 * LDS_STRIDE];             // A panel: 32 rows x K=1536 (~96.5KB)

    const int tid  = threadIdx.x;
    const int wave = tid >> 5;
    const int lane = tid & 31;
    const int rowPair = blockIdx.x & 3;
    const int jGroup  = blockIdx.x >> 2;
    const int m0 = rowPair * 32;
    const int j0 = jGroup * 64 + wave * 16;

    // ---- async stage of A panel (rows m0..m0+31, K 0..1535) into LDS ----
    // GLOBAL_LOAD_ASYNC_TO_LDS_B128: each lane copies 16B from global memory to its LDS address, counted by ASYNCcnt.
    const __bf16* xrow = xbf + (size_t)t * BATCH * DIM_IN;
    for (int idx = tid; idx < 32 * 192; idx += 128) {    // 192 chunks of 8 bf16 / row
        int row = idx / 192;
        int c8  = idx % 192;
        const __bf16* src = (c8 < 64)
            ? (xrow   + (size_t)(m0 + row) * DIM_IN + c8 * 8)          // K in [0,512)
            : (hprevB + (size_t)(m0 + row) * HID + (c8 - 64) * 8);     // K-512 in [0,1024)
        unsigned lds_off = (unsigned)(size_t)&ldsA[row * LDS_STRIDE + c8 * 8];
        asm volatile("global_load_async_to_lds_b128 %0, %1, off"
                     :: "v"(lds_off), "v"((unsigned long long)(size_t)src)
                     : "memory");
    }
    asm volatile("s_wait_asynccnt 0x0" ::: "memory");
    __syncthreads();

    // ---- fragment addressing (contiguous k = 16h + s per lane half, 16-bit operands, wave32) ----
    const int ln      = lane & 15;
    const int hiHalf  = lane >> 4;            // 0: lanes 0-15, 1: lanes 16-31
    const int ab_off  = hiHalf * 16;          // A: K base offset {0,16}
    const int kb_off  = hiHalf * 16;          // B: K base offset {0,16}

    const __bf16* arow0 = &ldsA[ln * LDS_STRIDE];          // M-tile 0: rows m0..m0+15
    const __bf16* arow1 = &ldsA[(16 + ln) * LDS_STRIDE];   // M-tile 1: rows m0+16..m0+31
    const __bf16* wr = Wt + (size_t)(j0 + ln) * KTOT + kb_off;
    const __bf16* wz = Wt + (size_t)(HID + j0 + ln) * KTOT + kb_off;
    const __bf16* wn = Wt + (size_t)(2 * HID + j0 + ln) * KTOT + kb_off;

    v8f cr0 = {}, cz0 = {}, cin0 = {}, chn0 = {};
    v8f cr1 = {}, cz1 = {}, cin1 = {}, chn1 = {};

    // ---- K = 0..511 : x-projection part (feeds r, z, and in) ----
    #pragma unroll 2
    for (int kk = 0; kk < DIM_IN; kk += 32) {
        Frag16 a0, a1, br, bz, bn;
        a0.q[0] = *(const uint4*)(arow0 + kk + ab_off);
        a0.q[1] = *(const uint4*)(arow0 + kk + ab_off + 8);
        a1.q[0] = *(const uint4*)(arow1 + kk + ab_off);
        a1.q[1] = *(const uint4*)(arow1 + kk + ab_off + 8);
        br.q[0] = *(const uint4*)(wr + kk);
        br.q[1] = *(const uint4*)(wr + kk + 8);
        bz.q[0] = *(const uint4*)(wz + kk);
        bz.q[1] = *(const uint4*)(wz + kk + 8);
        bn.q[0] = *(const uint4*)(wn + kk);
        bn.q[1] = *(const uint4*)(wn + kk + 8);
        cr0  = WMMA_BF16(a0.v, br.v, cr0);   cr1  = WMMA_BF16(a1.v, br.v, cr1);
        cz0  = WMMA_BF16(a0.v, bz.v, cz0);   cz1  = WMMA_BF16(a1.v, bz.v, cz1);
        cin0 = WMMA_BF16(a0.v, bn.v, cin0);  cin1 = WMMA_BF16(a1.v, bn.v, cin1);
    }
    // ---- K = 512..1535 : hidden part (feeds r, z, and hn) ----
    #pragma unroll 2
    for (int kk = DIM_IN; kk < KTOT; kk += 32) {
        Frag16 a0, a1, br, bz, bn;
        a0.q[0] = *(const uint4*)(arow0 + kk + ab_off);
        a0.q[1] = *(const uint4*)(arow0 + kk + ab_off + 8);
        a1.q[0] = *(const uint4*)(arow1 + kk + ab_off);
        a1.q[1] = *(const uint4*)(arow1 + kk + ab_off + 8);
        br.q[0] = *(const uint4*)(wr + kk);
        br.q[1] = *(const uint4*)(wr + kk + 8);
        bz.q[0] = *(const uint4*)(wz + kk);
        bz.q[1] = *(const uint4*)(wz + kk + 8);
        bn.q[0] = *(const uint4*)(wn + kk);
        bn.q[1] = *(const uint4*)(wn + kk + 8);
        cr0  = WMMA_BF16(a0.v, br.v, cr0);   cr1  = WMMA_BF16(a1.v, br.v, cr1);
        cz0  = WMMA_BF16(a0.v, bz.v, cz0);   cz1  = WMMA_BF16(a1.v, bz.v, cz1);
        chn0 = WMMA_BF16(a0.v, bn.v, chn0);  chn1 = WMMA_BF16(a1.v, bn.v, chn1);
    }

    // ---- epilogue: gates fused in registers ----
    // C/D layout: lanes 0-15 -> M = base+r, lanes 16-31 -> M = base+8+r; N = j0+ln.
    const int j = j0 + ln;
    const int mBase = m0 + hiHalf * 8;
    const float bbr = bi[j] + bh[j];
    const float bbz = bi[HID + j] + bh[HID + j];
    const float bin = bi[2 * HID + j];
    const float bhn = bh[2 * HID + j];

    #pragma unroll
    for (int r = 0; r < 8; ++r) {
        {   // M-tile 0
            const int m = mBase + r;
            const float hp = hprevF[(size_t)m * HID + j];
            const float rg = sigm(cr0[r] + bbr);
            const float zg = sigm(cz0[r] + bbz);
            const float ng = tanhf((cin0[r] + bin) + rg * (chn0[r] + bhn));
            const float hy = zg * ng + (1.0f - zg) * hp;   // reference's gate mix
            hnextF[(size_t)m * HID + j] = hy;
            hnextB[(size_t)m * HID + j] = (__bf16)hy;
        }
        {   // M-tile 1
            const int m = mBase + 16 + r;
            const float hp = hprevF[(size_t)m * HID + j];
            const float rg = sigm(cr1[r] + bbr);
            const float zg = sigm(cz1[r] + bbz);
            const float ng = tanhf((cin1[r] + bin) + rg * (chn1[r] + bhn));
            const float hy = zg * ng + (1.0f - zg) * hp;
            hnextF[(size_t)m * HID + j] = hy;
            hnextB[(size_t)m * HID + j] = (__bf16)hy;
        }
    }
}

// ---------------------------------------------------------------------------
extern "C" void kernel_launch(void* const* d_in, const int* in_sizes, int n_in,
                              void* d_out, int out_size, void* d_ws, size_t ws_size,
                              hipStream_t stream) {
    const float* x  = (const float*)d_in[0];
    const float* Wi = (const float*)d_in[1];
    const float* bi = (const float*)d_in[2];
    const float* Wh = (const float*)d_in[3];
    const float* bh = (const float*)d_in[4];
    float* out = (float*)d_out;

    char* ws = (char*)d_ws;
    size_t off = 0;
    __bf16* Wt  = (__bf16*)(ws + off); off += (size_t)NGATE * KTOT * 2;             // 9.4 MB
    __bf16* xbf = (__bf16*)(ws + off); off += (size_t)T_STEPS * BATCH * DIM_IN * 2; // 67 MB
    float*  hf0 = (float*)(ws + off);  off += (size_t)BATCH * HID * 4;
    float*  hf1 = (float*)(ws + off);  off += (size_t)BATCH * HID * 4;
    __bf16* hb0 = (__bf16*)(ws + off); off += (size_t)BATCH * HID * 2;
    __bf16* hb1 = (__bf16*)(ws + off); off += (size_t)BATCH * HID * 2;

    // one-time (per call) precision conversion; weights then live in L2 (192MB)
    build_wt <<<(NGATE * KTOT) / 256, 256, 0, stream>>>(Wi, Wh, Wt);
    convert_x<<<((size_t)T_STEPS * BATCH * DIM_IN) / 256, 256, 0, stream>>>(x, xbf);
    init_h   <<<(BATCH * HID) / 256, 256, 0, stream>>>(hf0, hb0);

    float*  hfs[2] = {hf0, hf1};
    __bf16* hbs[2] = {hb0, hb1};
    for (int t = 0; t < T_STEPS; ++t) {
        const float*  hpF = hfs[t & 1];
        const __bf16* hpB = hbs[t & 1];
        float*  hnF = (t == T_STEPS - 1) ? out : hfs[1 - (t & 1)];
        __bf16* hnB = hbs[1 - (t & 1)];
        gru_step<<<64, 128, 0, stream>>>(xbf, Wt, bi, bh, hpF, hpB, hnF, hnB, t);
    }
}
